// MaskedMultiHeadedSelfAttention_37847251813048
// MI455X (gfx1250) — matrix operand order repaired and checked by reading, not compiled, not run on hardware
//
#include <hip/hip_runtime.h>

typedef __bf16 bf16_t;
typedef __attribute__((ext_vector_type(16))) __bf16 v16bf;
typedef __attribute__((ext_vector_type(8)))  __bf16 v8bf;
typedef __attribute__((ext_vector_type(8)))  float  v8f;
typedef unsigned int u32x4 __attribute__((ext_vector_type(4)));
typedef int i32x4 __attribute__((ext_vector_type(4)));
typedef int i32x8 __attribute__((ext_vector_type(8)));

#define B_ 4
#define S_ 2048
#define E_ 768
#define H_ 12
#define D_ 64

static __device__ __forceinline__ v8f vzero() {
  v8f z = {0.f, 0.f, 0.f, 0.f, 0.f, 0.f, 0.f, 0.f};
  return z;
}

static __device__ __forceinline__ v8f wmma_bf16(v16bf a, v16bf b, v8f c) {
  // D = A(16x32 bf16) * B(32x16 bf16) + C(16x16 f32)
  return __builtin_amdgcn_wmma_f32_16x16x32_bf16(false, a, false, b, (short)0, c,
                                                 false, false);
}

static __device__ __forceinline__ v16bf make_v16(v8bf lo, v8bf hi) {
  return __builtin_shufflevector(lo, hi, 0, 1, 2, 3, 4, 5, 6, 7, 8, 9, 10, 11,
                                 12, 13, 14, 15);
}

// A fragment (16x32 bf16). base points at row 0 of the 16-row strip, row-major,
// leading dim ld (elements). Lane l<16 holds row l, K = kBase+0..15;
// lane l>=16 holds row l-16, K = kBase+16..31.
static __device__ __forceinline__ v16bf load_a_frag(const bf16_t* base, int ld,
                                                    int kBase, int lane) {
  const int r = lane & 15;
  const int kOff = (lane & 16) ? 16 : 0;
  const bf16_t* p = base + (size_t)r * ld + kBase + kOff;
  v8bf lo = *(const v8bf*)p;
  v8bf hi = *(const v8bf*)(p + 8);
  return make_v16(lo, hi);
}

// B fragment (32x16 bf16) where element (k, n) lives at colBase[n*ldn + k]
// (K contiguous per column). Lanes 0-15: col n, K=kBase+0..15;
// lanes 16-31: col n-16, K=kBase+16..31.
static __device__ __forceinline__ v16bf load_b_frag_ncol(const bf16_t* colBase,
                                                         size_t ldn, int kBase,
                                                         int lane) {
  const int n = lane & 15;
  const int kOff = (lane & 16) ? 16 : 0;
  return *(const v16bf*)(colBase + (size_t)n * ldn + kBase + kOff);
}

// B fragment from pre-packed weights: fragments stored as contiguous 512-elem
// blocks in (kTile, nTile) order, 16 bf16 per lane.
static __device__ __forceinline__ v16bf load_b_frag_packed(const bf16_t* wp,
                                                           int kTile, int nTile,
                                                           int NT, int lane) {
  return *(const v16bf*)(wp + (((size_t)(kTile * NT + nTile)) * 32 + lane) * 16);
}

// ---------------------------------------------------------------------------
// Tensor Data Mover: issue a <=2D tensor tile load into LDS.
// D# packing per CDNA5 ISA (group0 128b, group1 256b):
//   g0: [1:0]=count=1, [63:32]=lds_addr, [120:64]=global_addr, [127:126]=type=2
//   g1: [17:16]=data_size, [79:48]=tensor_dim0, [111:80]=tensor_dim1,
//       [127:112]=tile_dim0, [143:128]=tile_dim1, [207:160]=tensor_dim0_stride
static __device__ __forceinline__ void tdm_load_to_lds(
    unsigned lds_off, unsigned long long gaddr, unsigned dsz_code,
    unsigned tile0, unsigned tile1, unsigned tensor0, unsigned tensor1,
    unsigned long long stride0) {
  u32x4 g0;
  g0.x = 1u;                                            // count=1, user mode
  g0.y = lds_off;                                       // lds_addr (bytes)
  g0.z = (unsigned)(gaddr & 0xFFFFFFFFull);             // global_addr[31:0]
  g0.w = (unsigned)((gaddr >> 32) & 0x01FFFFFFu) | (2u << 30);  // type=2
  i32x8 g1;
  g1[0] = (int)(dsz_code << 16);                        // wg_mask=0, data_size
  g1[1] = (int)((tensor0 & 0xFFFFu) << 16);             // tensor_dim0[15:0]
  g1[2] = (int)(((tensor0 >> 16) & 0xFFFFu) | ((tensor1 & 0xFFFFu) << 16));
  g1[3] = (int)(((tensor1 >> 16) & 0xFFFFu) | ((tile0 & 0xFFFFu) << 16));
  g1[4] = (int)(tile1 & 0xFFFFu);                       // tile_dim2 = 0
  g1[5] = (int)(stride0 & 0xFFFFFFFFull);               // dim0_stride[31:0]
  g1[6] = (int)((stride0 >> 32) & 0xFFFFull);           // dim0_stride[47:32]
  g1[7] = 0;
  i32x4 gz = {0, 0, 0, 0};
#if __clang_major__ >= 23
  i32x8 gz8 = {0, 0, 0, 0, 0, 0, 0, 0};
  __builtin_amdgcn_tensor_load_to_lds(g0, g1, gz, gz, gz8, 0);
#else
  __builtin_amdgcn_tensor_load_to_lds(g0, g1, gz, gz, 0);
#endif
}

// ---------------------------------------------------------------------------
// Kernel 0a: x f32 -> bf16
__global__ void cvt_f32_bf16_kernel(const float* __restrict__ src,
                                    bf16_t* __restrict__ dst, int n) {
  int i = blockIdx.x * 256 + threadIdx.x;
  if (i < n) dst[i] = (bf16_t)src[i];
}

// Kernel 0b: pack Wq/Wk/Wv (H,E,D) f32 -> B-fragment-ordered bf16 [3][H][...]
__global__ void pack_wqkv_kernel(const float* __restrict__ Wq,
                                 const float* __restrict__ Wk,
                                 const float* __restrict__ Wv,
                                 bf16_t* __restrict__ dst) {
  const int per = H_ * E_ * D_;
  int i = blockIdx.x * 256 + threadIdx.x;
  if (i >= 3 * per) return;
  const int w = i / per;
  int rem = i - w * per;
  const int h = rem / (E_ * D_);
  rem -= h * (E_ * D_);
  const int k = rem / D_;
  const int n = rem - k * D_;
  const float* src = (w == 0) ? Wq : (w == 1) ? Wk : Wv;
  const float v = src[((size_t)h * E_ + k) * D_ + n];
  const int kTile = k >> 5, kIn = k & 31, nTile = n >> 4, nIn = n & 15;
  const int lane = nIn + ((kIn >> 4) << 4);
  const int e = kIn & 15;
  dst[((size_t)(w * H_ + h)) * (E_ * D_) +
      (((size_t)(kTile * (D_ / 16) + nTile)) * 32 + lane) * 16 + e] = (bf16_t)v;
}

// Kernel 0c: pack Wo (E, E) f32 -> B-fragment-ordered bf16
__global__ void pack_wo_kernel(const float* __restrict__ Wo,
                               bf16_t* __restrict__ dst) {
  int i = blockIdx.x * 256 + threadIdx.x;
  if (i >= E_ * E_) return;
  const int k = i / E_;
  const int n = i - k * E_;
  const float v = Wo[(size_t)k * E_ + n];
  const int kTile = k >> 5, kIn = k & 31, nTile = n >> 4, nIn = n & 15;
  const int lane = nIn + ((kIn >> 4) << 4);
  const int e = kIn & 15;
  dst[(((size_t)(kTile * (E_ / 16) + nTile)) * 32 + lane) * 16 + e] = (bf16_t)v;
}

// ---------------------------------------------------------------------------
// Kernel 1: QKV projection. grid = (S/64, B, 3*H), block = 128 (4 waves).
// Wave computes a 16x64 strip. Q is pre-scaled by 1/sqrt(D); V is stored
// transposed (D x S) so PV B-fragments are K-contiguous.
__global__ __launch_bounds__(128) void qkv_proj_kernel(
    const bf16_t* __restrict__ xbf, const bf16_t* __restrict__ wpack,
    const float* __restrict__ bq, const float* __restrict__ bk,
    const float* __restrict__ bv, bf16_t* __restrict__ Qo,
    bf16_t* __restrict__ Ko, bf16_t* __restrict__ Vto) {
  const int sTile = blockIdx.x, b = blockIdx.y, z = blockIdx.z;
  const int w = z % 3, h = z / 3;
  const int tid = threadIdx.x, lane = tid & 31, wv = tid >> 5;
  const int sBase = sTile * 64 + wv * 16;
  const bf16_t* xb = xbf + ((size_t)b * S_ + sBase) * E_;
  const bf16_t* wp = wpack + ((size_t)(w * H_ + h)) * (E_ * D_);

  v8f acc[4];
#pragma unroll
  for (int nt = 0; nt < 4; ++nt) acc[nt] = vzero();

  for (int kt = 0; kt < E_ / 32; ++kt) {
    v16bf af = load_a_frag(xb, E_, kt * 32, lane);
#pragma unroll
    for (int nt = 0; nt < 4; ++nt)
      acc[nt] = wmma_bf16(af, load_b_frag_packed(wp, kt, nt, D_ / 16, lane),
                          acc[nt]);
  }

  const int n0 = lane & 15, half = (lane >> 4) & 1;
  const float* bias = (w == 0) ? bq : (w == 1) ? bk : bv;
  if (w < 2) {
    bf16_t* dst = ((w == 0) ? Qo : Ko) + ((size_t)(b * H_ + h)) * S_ * D_;
    const float scale = (w == 0) ? 0.125f : 1.0f;  // 1/sqrt(64) folded into Q
#pragma unroll
    for (int nt = 0; nt < 4; ++nt) {
      const int d = nt * 16 + n0;
      const float bb = bias[h * D_ + d];
#pragma unroll
      for (int r = 0; r < 8; ++r) {
        const int s = sBase + r + 8 * half;
        dst[(size_t)s * D_ + d] = (bf16_t)((acc[nt][r] + bb) * scale);
      }
    }
  } else {
    bf16_t* dst = Vto + ((size_t)(b * H_ + h)) * D_ * S_;  // transposed (D,S)
#pragma unroll
    for (int nt = 0; nt < 4; ++nt) {
      const int d = nt * 16 + n0;
      const float bb = bias[h * D_ + d];
      v8bf pk;
#pragma unroll
      for (int r = 0; r < 8; ++r) pk[r] = (bf16_t)(acc[nt][r] + bb);
      *(v8bf*)(dst + (size_t)d * S_ + sBase + 8 * half) = pk;  // 8 consec s
    }
  }
}

// ---------------------------------------------------------------------------
// Kernel 2: flash attention with TDM-staged K/V tiles.
// grid = (S/64, H, B), block = 128 (4 waves). Wave 0 drives the Tensor Data
// Mover: double-buffered 32-key K (32x64, contiguous 4KB) and V-transposed
// (64 d-rows x 32 keys, stride S) tiles into LDS, s_wait_tensorcnt + barrier
// publishes them, all 4 waves consume B-fragments from LDS (4x less VMEM).
__global__ __launch_bounds__(128) void attn_flash_kernel(
    const bf16_t* __restrict__ Q, const bf16_t* __restrict__ K,
    const bf16_t* __restrict__ Vt, const int* __restrict__ mask,
    bf16_t* __restrict__ AO) {
  struct SMem {
    float maskb[S_];                         // 8 KB
    bf16_t kbuf[2][32 * 64];                 // 2 x 4 KB, row-major 32x64
    bf16_t vbuf[2][64 * 32];                 // 2 x 4 KB, [d][key] ld=32
    bf16_t pbuf[4][16 * 32];                 // 4 KB, per-wave P tiles
  };
  __shared__ __align__(32) SMem sm;
  const int b = blockIdx.z, h = blockIdx.y, qb = blockIdx.x;
  const int tid = threadIdx.x, lane = tid & 31, wv = tid >> 5;

  for (int i = tid; i < S_; i += 128)
    sm.maskb[i] = (mask[b * S_ + i] != 0) ? 0.0f : -1.0e9f;

  const bf16_t* Qb = Q + ((size_t)(b * H_ + h)) * S_ * D_;
  const bf16_t* Kb = K + ((size_t)(b * H_ + h)) * S_ * D_;
  const bf16_t* Vb = Vt + ((size_t)(b * H_ + h)) * D_ * S_;
  const int sBase = qb * 64 + wv * 16;
  const int n0 = lane & 15, half = (lane >> 4) & 1;
  bf16_t* pb = &sm.pbuf[wv][0];
  // Static LDS is allocated from offset 0; byte offsets for the TDM D#.
  const unsigned koff[2] = {(unsigned)((char*)&sm.kbuf[0][0] - (char*)&sm),
                            (unsigned)((char*)&sm.kbuf[1][0] - (char*)&sm)};
  const unsigned voff[2] = {(unsigned)((char*)&sm.vbuf[0][0] - (char*)&sm),
                            (unsigned)((char*)&sm.vbuf[1][0] - (char*)&sm)};

  const v16bf qf0 = load_a_frag(Qb + (size_t)sBase * D_, D_, 0, lane);
  const v16bf qf1 = load_a_frag(Qb + (size_t)sBase * D_, D_, 32, lane);

  v8f oacc0 = vzero(), oacc1 = vzero(), oacc2 = vzero(), oacc3 = vzero();
  float mrun[8], lrun[8];
#pragma unroll
  for (int r = 0; r < 8; ++r) { mrun[r] = -1.0e30f; lrun[r] = 0.0f; }

  const int NB = S_ / 32;
  if (wv == 0) {
    // K block 0: 32 consecutive rows = 2048 contiguous bf16 (1-D tile).
    tdm_load_to_lds(koff[0], (unsigned long long)(size_t)Kb, 1u, 2048u, 0u,
                    (unsigned)(S_ * D_), 1u, (unsigned long long)(S_ * D_));
    // V block 0: 64 rows of 32 bf16, row stride S_.
    tdm_load_to_lds(voff[0], (unsigned long long)(size_t)Vb, 1u, 32u, 64u,
                    (unsigned)S_, (unsigned)D_, (unsigned long long)S_);
  }

  for (int j = 0; j < NB; ++j) {
    const int kb = j * 32;
    if (wv == 0) __builtin_amdgcn_s_wait_tensorcnt(0);
    __syncthreads();  // publish TDM-written LDS tile (and maskb on j==0)
    if (wv == 0 && j + 1 < NB) {
      const int nb = (j + 1) & 1;
      tdm_load_to_lds(koff[nb],
                      (unsigned long long)(size_t)(Kb + (size_t)(kb + 32) * D_),
                      1u, 2048u, 0u, (unsigned)(S_ * D_), 1u,
                      (unsigned long long)(S_ * D_));
      tdm_load_to_lds(voff[nb], (unsigned long long)(size_t)(Vb + kb + 32), 1u,
                      32u, 64u, (unsigned)S_, (unsigned)D_,
                      (unsigned long long)S_);
    }
    const bf16_t* kl = &sm.kbuf[j & 1][0];
    const bf16_t* vl = &sm.vbuf[j & 1][0];

    // scores: Q(16x64) * K^T(64x32keys) -> two 16x16 f32 frags (from LDS)
    v16bf bk00 = load_b_frag_ncol(kl, D_, 0, lane);
    v16bf bk01 = load_b_frag_ncol(kl, D_, 32, lane);
    v16bf bk10 = load_b_frag_ncol(kl + 16 * D_, D_, 0, lane);
    v16bf bk11 = load_b_frag_ncol(kl + 16 * D_, D_, 32, lane);
    v8f s0 = vzero(), s1 = vzero();
    s0 = wmma_bf16(qf0, bk00, s0);
    s0 = wmma_bf16(qf1, bk01, s0);
    s1 = wmma_bf16(qf0, bk10, s1);
    s1 = wmma_bf16(qf1, bk11, s1);

    const float bias0 = sm.maskb[kb + n0];
    const float bias1 = sm.maskb[kb + 16 + n0];
    float alpha[8];
#pragma unroll
    for (int r = 0; r < 8; ++r) {
      const float v0 = s0[r] + bias0;
      const float v1 = s1[r] + bias1;
      float mx = fmaxf(v0, v1);  // row stats live in 16-lane halves
      mx = fmaxf(mx, __shfl_xor(mx, 1, 32));
      mx = fmaxf(mx, __shfl_xor(mx, 2, 32));
      mx = fmaxf(mx, __shfl_xor(mx, 4, 32));
      mx = fmaxf(mx, __shfl_xor(mx, 8, 32));
      const float mnew = fmaxf(mrun[r], mx);
      alpha[r] = __expf(mrun[r] - mnew);
      mrun[r] = mnew;
      const float p0 = __expf(v0 - mnew);
      const float p1 = __expf(v1 - mnew);
      float sum = p0 + p1;
      sum += __shfl_xor(sum, 1, 32);
      sum += __shfl_xor(sum, 2, 32);
      sum += __shfl_xor(sum, 4, 32);
      sum += __shfl_xor(sum, 8, 32);
      lrun[r] = lrun[r] * alpha[r] + sum;
      // scatter P to per-wave LDS tile (C layout -> row-major 16x32)
      pb[(r + 8 * half) * 32 + n0] = (bf16_t)p0;
      pb[(r + 8 * half) * 32 + 16 + n0] = (bf16_t)p1;
    }
#pragma unroll
    for (int r = 0; r < 8; ++r) {
      oacc0[r] *= alpha[r];
      oacc1[r] *= alpha[r];
      oacc2[r] *= alpha[r];
      oacc3[r] *= alpha[r];
    }
    // Per-wave LDS is in-order (DScnt); fence only stops compiler reordering.
    __builtin_amdgcn_wave_barrier();
    v16bf pA;
    {
      const int rr = lane & 15;
      const int kOff = (lane & 16) ? 16 : 0;
      const bf16_t* p = pb + rr * 32 + kOff;
      v8bf lo = *(const v8bf*)p;
      v8bf hi = *(const v8bf*)(p + 8);
      pA = make_v16(lo, hi);
    }
    __builtin_amdgcn_wave_barrier();
    // O += P(16x32) * V(32keys x 64d), V tile in LDS as [d][key], ld=32
    oacc0 = wmma_bf16(pA, load_b_frag_ncol(vl + 0 * 32, 32, 0, lane), oacc0);
    oacc1 = wmma_bf16(pA, load_b_frag_ncol(vl + 16 * 32, 32, 0, lane), oacc1);
    oacc2 = wmma_bf16(pA, load_b_frag_ncol(vl + 32 * 32, 32, 0, lane), oacc2);
    oacc3 = wmma_bf16(pA, load_b_frag_ncol(vl + 48 * 32, 32, 0, lane), oacc3);
  }

#pragma unroll
  for (int r = 0; r < 8; ++r) {
    const int s = sBase + r + 8 * half;
    const float inv = 1.0f / lrun[r];
    bf16_t* dst = AO + ((size_t)(b * S_ + s)) * (H_ * D_) + h * D_;
    dst[0 + n0] = (bf16_t)(oacc0[r] * inv);
    dst[16 + n0] = (bf16_t)(oacc1[r] * inv);
    dst[32 + n0] = (bf16_t)(oacc2[r] * inv);
    dst[48 + n0] = (bf16_t)(oacc3[r] * inv);
  }
}

// ---------------------------------------------------------------------------
// Kernel 3: output projection (B*S, 768) x (768, 768) + bo -> f32 out.
// grid = (B*S/64, E/64), block = 128 (4 waves), wave = 16x64 strip.
__global__ __launch_bounds__(128) void out_proj_kernel(
    const bf16_t* __restrict__ AO, const bf16_t* __restrict__ wopack,
    const float* __restrict__ bo, float* __restrict__ out) {
  const int mTile = blockIdx.x, nTile64 = blockIdx.y;
  const int tid = threadIdx.x, lane = tid & 31, wv = tid >> 5;
  const int mBase = mTile * 64 + wv * 16;
  const bf16_t* ab = AO + (size_t)mBase * E_;

  v8f acc[4];
#pragma unroll
  for (int nt = 0; nt < 4; ++nt) acc[nt] = vzero();

  for (int kt = 0; kt < E_ / 32; ++kt) {
    v16bf af = load_a_frag(ab, E_, kt * 32, lane);
#pragma unroll
    for (int nt = 0; nt < 4; ++nt)
      acc[nt] = wmma_bf16(
          af, load_b_frag_packed(wopack, kt, nTile64 * 4 + nt, E_ / 16, lane),
          acc[nt]);
  }

  const int n0 = lane & 15, half = (lane >> 4) & 1;
#pragma unroll
  for (int nt = 0; nt < 4; ++nt) {
    const int n = nTile64 * 64 + nt * 16 + n0;
    const float bb = bo[n];
#pragma unroll
    for (int r = 0; r < 8; ++r) {
      const int m = mBase + r + 8 * half;
      out[(size_t)m * E_ + n] = acc[nt][r] + bb;
    }
  }
}

// ---------------------------------------------------------------------------
extern "C" void kernel_launch(void* const* d_in, const int* in_sizes, int n_in,
                              void* d_out, int out_size, void* d_ws,
                              size_t ws_size, hipStream_t stream) {
  (void)in_sizes; (void)n_in; (void)out_size; (void)ws_size;
  const float* x = (const float*)d_in[0];
  const int* mask = (const int*)d_in[1];
  const float* Wq = (const float*)d_in[2];
  const float* bq = (const float*)d_in[3];
  const float* Wk = (const float*)d_in[4];
  const float* bk = (const float*)d_in[5];
  const float* Wv = (const float*)d_in[6];
  const float* bv = (const float*)d_in[7];
  const float* Wo = (const float*)d_in[8];
  const float* bo = (const float*)d_in[9];
  float* out = (float*)d_out;

  char* ws = (char*)d_ws;
  size_t off = 0;
  auto take = [&](size_t bytes) -> char* {
    char* p = ws + off;
    off += (bytes + 255) & ~(size_t)255;
    return p;
  };
  bf16_t* xbf = (bf16_t*)take((size_t)B_ * S_ * E_ * 2);          // 12.6 MB
  bf16_t* wqkv = (bf16_t*)take((size_t)3 * H_ * E_ * D_ * 2);     //  3.5 MB
  bf16_t* wop = (bf16_t*)take((size_t)E_ * E_ * 2);               //  1.2 MB
  bf16_t* Qb = (bf16_t*)take((size_t)B_ * H_ * S_ * D_ * 2);      // 12.6 MB
  bf16_t* Kb = (bf16_t*)take((size_t)B_ * H_ * S_ * D_ * 2);      // 12.6 MB
  bf16_t* Vt = (bf16_t*)take((size_t)B_ * H_ * D_ * S_ * 2);      // 12.6 MB
  bf16_t* AO = (bf16_t*)take((size_t)B_ * S_ * H_ * D_ * 2);      // 12.6 MB

  const int nX = B_ * S_ * E_;
  cvt_f32_bf16_kernel<<<(nX + 255) / 256, 256, 0, stream>>>(x, xbf, nX);
  const int nWqkv = 3 * H_ * E_ * D_;
  pack_wqkv_kernel<<<(nWqkv + 255) / 256, 256, 0, stream>>>(Wq, Wk, Wv, wqkv);
  const int nWo = E_ * E_;
  pack_wo_kernel<<<(nWo + 255) / 256, 256, 0, stream>>>(Wo, wop);

  qkv_proj_kernel<<<dim3(S_ / 64, B_, 3 * H_), 128, 0, stream>>>(
      xbf, wqkv, bq, bk, bv, Qb, Kb, Vt);
  attn_flash_kernel<<<dim3(S_ / 64, H_, B_), 128, 0, stream>>>(Qb, Kb, Vt, mask,
                                                               AO);
  out_proj_kernel<<<dim3(B_ * S_ / 64, E_ / 64), 128, 0, stream>>>(AO, wop, bo,
                                                                   out);
}
